// softCorrelationReverseTransform_38929583571143
// MI455X (gfx1250) — matrix operand order repaired and checked by reading, not compiled, not run on hardware
//
#include <hip/hip_runtime.h>

// ---------------------------------------------------------------------------
// Expected-value softmax decode:  out[i] = softmax(x[i,:] @ di) . arange(256)
//   x : [N=524288, 64] fp32     di : [64, 256] fp32     out : [N] fp32
//
// Strategy: f16 WMMA (v_wmma_f32_16x16x32_f16) for the GEMM, fp32 accum,
// di pre-swizzled to WMMA B-fragment layout in LDS once per block,
// all 256 logits per 16-row tile held in 16 v8f accumulators,
// softmax via lane-local + shfl_xor reductions (wave32 halves).
// ---------------------------------------------------------------------------

typedef _Float16 v16h __attribute__((ext_vector_type(16)));
typedef float    v8f  __attribute__((ext_vector_type(8)));

#define DDIM   64      // K dimension
#define VRANGE 256     // number of logits / columns
#define TILES  16      // VRANGE / 16 column tiles
#define WAVES  8       // waves per block (256 threads, wave32)
#define BLOCK  256
#define ROWS_PER_CHUNK 128   // WAVES * 16 rows

__global__ __launch_bounds__(BLOCK)
void ev_softmax_wmma_kernel(const float* __restrict__ x,
                            const float* __restrict__ di,
                            float* __restrict__ out,
                            int nrows)
{
    // 32 B-fragments: fragment f = tile*2 + kstep, each = 32 lanes x 16 halfs.
    // Per-lane layout used here (16-bit B 32x16, contiguous k, same as A): col = 16*t + (lane&15),
    // k = 32*s + (lane<16 ? 0 : 16) + j, j = 0..15 contiguous.
    __shared__ __align__(32) _Float16 Bsh[32 * 512];   // 32 KB

    const int tid    = threadIdx.x;
    const int lane   = tid & 31;
    const int wave   = tid >> 5;
    const int half16 = lane & 15;
    const int hi     = lane >> 4;          // 0 -> C rows 0..7, 1 -> rows 8..15

    // ---------------- stage di into LDS in B-fragment layout (fp32 -> f16) --
    for (int slot = tid; slot < 32 * 32; slot += BLOCK) {
        const int f = slot >> 5;           // fragment 0..31
        const int l = slot & 31;           // lane within fragment
        const int t = f >> 1, s = f & 1;
        const int col   = (t << 4) + (l & 15);
        const int kbase = (s << 5) + ((l < 16) ? 0 : 16);
        _Float16* dst = &Bsh[f * 512 + l * 16];
        #pragma unroll
        for (int j = 0; j < 16; ++j)
            dst[j] = (_Float16)di[(kbase + j) * VRANGE + col];
    }
    __syncthreads();

    const int nchunks = nrows / ROWS_PER_CHUNK;   // N is a multiple of 128

    for (int c = blockIdx.x; c < nchunks; c += gridDim.x) {
        const int row_base = (c * WAVES + wave) << 4;   // 16 rows per wave
        const int row      = row_base + half16;
        const float* xr    = x + (size_t)row * DDIM;

        // prefetch next chunk's rows for this wave (global_prefetch_b8)
        {
            int nc = c + gridDim.x;
            if (nc < nchunks) {
                const float* nx = x + (size_t)(((nc * WAVES + wave) << 4) + half16) * DDIM;
                __builtin_prefetch(nx, 0, 3);
            }
        }

        // ---------------- A fragments (16-bit A 16x32, contiguous k order) ----------
        // lanes 0-15 (row M=lane): j0..15 -> k = 32s + j
        // lanes 16-31 (same M):    j0..15 -> k = 32s+16+j
        v16h a0, a1;
        #pragma unroll
        for (int s = 0; s < 2; ++s) {
            const int c0 = (s << 5) + (hi ? 16 : 0);
            const float4 q0 = ((const float4*)(xr + c0))[0];
            const float4 q1 = ((const float4*)(xr + c0))[1];
            const float4 q2 = ((const float4*)(xr + c0))[2];
            const float4 q3 = ((const float4*)(xr + c0))[3];
            const float tmp[16] = { q0.x,q0.y,q0.z,q0.w, q1.x,q1.y,q1.z,q1.w,
                                    q2.x,q2.y,q2.z,q2.w, q3.x,q3.y,q3.z,q3.w };
            v16h a;
            #pragma unroll
            for (int j = 0; j < 16; ++j) a[j] = (_Float16)tmp[j];
            if (s == 0) a0 = a; else a1 = a;
        }

        // ---------------- GEMM: 16 column tiles, K = 64 in two WMMA steps ---
        v8f acc[TILES];
        #pragma unroll
        for (int t = 0; t < TILES; ++t) {
            const v16h b0 = *(const v16h*)&Bsh[(t * 2 + 0) * 512 + lane * 16];
            const v16h b1 = *(const v16h*)&Bsh[(t * 2 + 1) * 512 + lane * 16];
            v8f z = {};
            z = __builtin_amdgcn_wmma_f32_16x16x32_f16(false, a0, false, b0,
                                                       (short)0, z, false, false);
            z = __builtin_amdgcn_wmma_f32_16x16x32_f16(false, a1, false, b1,
                                                       (short)0, z, false, false);
            acc[t] = z;
        }

        // ---------------- softmax + expected value --------------------------
        // C layout: lane holds column n = 16*t + half16; VGPR slot r holds
        // row (8*hi + r). Reduce across columns = across tiles (local) then
        // across the 16 lanes of each half (shfl_xor masks 1,2,4,8).
        float m[8];
        #pragma unroll
        for (int r = 0; r < 8; ++r) m[r] = acc[0][r];
        #pragma unroll
        for (int t = 1; t < TILES; ++t)
            #pragma unroll
            for (int r = 0; r < 8; ++r) m[r] = fmaxf(m[r], acc[t][r]);
        #pragma unroll
        for (int sh = 1; sh <= 8; sh <<= 1)
            #pragma unroll
            for (int r = 0; r < 8; ++r)
                m[r] = fmaxf(m[r], __shfl_xor(m[r], sh, 32));

        float ssum[8], wsum[8];
        #pragma unroll
        for (int r = 0; r < 8; ++r) { ssum[r] = 0.f; wsum[r] = 0.f; }
        #pragma unroll
        for (int t = 0; t < TILES; ++t) {
            const float cn = (float)((t << 4) + half16);
            #pragma unroll
            for (int r = 0; r < 8; ++r) {
                const float e = __expf(acc[t][r] - m[r]);
                ssum[r] += e;
                wsum[r] += e * cn;
            }
        }
        #pragma unroll
        for (int sh = 1; sh <= 8; sh <<= 1)
            #pragma unroll
            for (int r = 0; r < 8; ++r) {
                ssum[r] += __shfl_xor(ssum[r], sh, 32);
                wsum[r] += __shfl_xor(wsum[r], sh, 32);
            }

        // 16 lanes write 16 consecutive rows: lane (hi,half16<8) owns slot half16
        if (half16 < 8) {
            const int r = half16;
            out[row_base + hi * 8 + r] = wsum[r] / ssum[r];
        }
    }
}

extern "C" void kernel_launch(void* const* d_in, const int* in_sizes, int n_in,
                              void* d_out, int out_size, void* d_ws, size_t ws_size,
                              hipStream_t stream)
{
    const float* x  = (const float*)d_in[0];   // [N, 64] fp32
    const float* di = (const float*)d_in[1];   // [64, 256] fp32
    float* out      = (float*)d_out;           // [N] fp32

    const int nrows   = in_sizes[0] / DDIM;            // 524288
    const int nchunks = nrows / ROWS_PER_CHUNK;        // 4096
    int grid = nchunks < 2048 ? nchunks : 2048;        // grid-stride over chunks
    if (grid < 1) grid = 1;

    ev_softmax_wmma_kernel<<<dim3(grid), dim3(BLOCK), 0, stream>>>(x, di, out, nrows);
}
